// GraphSAGE_36979668418994
// MI455X (gfx1250) — compile-verified, matrix operand order checked by reading, not run on hardware
//
#include <hip/hip_runtime.h>
#include <hip/hip_bf16.h>

// ---------------- problem constants ----------------
#define N_NODES 100000
#define N_EDGES 1000000
#define F_IN    256
#define HID     256
#define HID2    128
#define K_FUSED 512   // [mean | self] concatenated K for both layers

typedef __attribute__((ext_vector_type(16))) __bf16 v16bf;
typedef __attribute__((ext_vector_type(8)))  float  v8f;

// ---------------- workspace layout (float units) ----------------
// cnt:   [N_NODES]
// dinv:  [N_NODES]
// agg:   [N_NODES * 256]
// h1:    [N_NODES * 256]
// h2:    [N_NODES * 128]
// W1t:   bf16 [256][512]  (as 65536 floats)
// W2t:   bf16 [128][512]  (as 32768 floats)
#define O_CNT  ((size_t)0)
#define O_DINV ((size_t)N_NODES)
#define O_AGG  ((size_t)2 * N_NODES)
#define O_H1   (O_AGG + (size_t)N_NODES * 256)
#define O_H2   (O_H1  + (size_t)N_NODES * 256)
#define O_W1T  (O_H2  + (size_t)N_NODES * 128)
#define O_W2T  (O_W1T + (size_t)(256 * 512) / 2)

// ---------------- degree ----------------
__global__ void deg_count_k(const int* __restrict__ dst, float* __restrict__ cnt) {
    int e = blockIdx.x * 256 + threadIdx.x;
    if (e < N_EDGES) atomicAdd(&cnt[dst[e]], 1.0f);
}

__global__ void deg_inv_k(const float* __restrict__ cnt, float* __restrict__ dinv) {
    int n = blockIdx.x * 256 + threadIdx.x;
    if (n < N_NODES) {
        float c = cnt[n];
        dinv[n] = (c > 0.0f) ? (1.0f / c) : 0.0f;
    }
}

// ---------------- weight prep: fused transpose + bf16 convert ----------------
// W1t[n][k] = k<256 ? W1l[k][n] : W1r[k-256][n]   (W1l/W1r row-major [256][256])
// W2t[n][k] = k<256 ? W2l[k][n] : W2r[k-256][n]   (W2l/W2r row-major [256][128])
__global__ void prep_w_k(const float* __restrict__ W1l, const float* __restrict__ W1r,
                         const float* __restrict__ W2l, const float* __restrict__ W2r,
                         __bf16* __restrict__ W1t, __bf16* __restrict__ W2t) {
    int t = blockIdx.x * 256 + threadIdx.x;
    if (t < 256 * 512) {
        int n = t >> 9, k = t & 511;
        float v = (k < 256) ? W1l[k * 256 + n] : W1r[(k - 256) * 256 + n];
        W1t[(size_t)n * 512 + k] = (__bf16)v;
    } else {
        int t2 = t - 256 * 512;   // grid sized exactly, t2 < 128*512
        int n = t2 >> 9, k = t2 & 511;
        float v = (k < 256) ? W2l[k * 128 + n] : W2r[(k - 256) * 128 + n];
        W2t[(size_t)n * 512 + k] = (__bf16)v;
    }
}

// ---------------- edge scatter-add (mean numerator) ----------------
// 256 threads = 4 edges x 64 lanes; each lane owns a float4 feature chunk.
__global__ void scatter_add_k(const float* __restrict__ feat,
                              const int* __restrict__ src,
                              const int* __restrict__ dst,
                              float* __restrict__ agg) {
    int t = blockIdx.x * 256 + threadIdx.x;
    int e = t >> 6;
    int lane = t & 63;
    if (e >= N_EDGES) return;
    int s = src[e];
    int d = dst[e];
    if (e + 4 < N_EDGES)  // pull next edge's source row toward the caches
        __builtin_prefetch(feat + (size_t)src[e + 4] * 256 + lane * 4, 0, 1);
    const float4 v = *(const float4*)(feat + (size_t)s * 256 + lane * 4);
    float* o = agg + (size_t)d * 256 + lane * 4;
    atomicAdd(o + 0, v.x);
    atomicAdd(o + 1, v.y);
    atomicAdd(o + 2, v.z);
    atomicAdd(o + 3, v.w);
}

// ---------------- fused SAGE layer GEMM ----------------
// out[m][n] = relu( sum_k A[m][k] * Bt[n][k] + bias[n] )
//   A[m][k] = k<256 ? agg[m][k]*dinv[m] : self[m][k-256]
// One wave computes a 16(M) x 64(N) strip; NG waves/block cover N = 64*NG.
// grid.x = M/16 (M = 100000 = 6250*16 exactly -> no guards, EXEC all ones).
template <int NG>
__global__ void sage_gemm_k(const float* __restrict__ agg,
                            const float* __restrict__ self,
                            const float* __restrict__ dinv,
                            const __bf16* __restrict__ Bt,
                            const float* __restrict__ bias,
                            float* __restrict__ out, int N) {
    const int wave = threadIdx.x >> 5;
    const int lane = threadIdx.x & 31;
    const int l15  = lane & 15;
    const int half = lane >> 4;           // 0: K 0..15, 1: K 16..31 of the step
    const int mbase = blockIdx.x * 16;
    const int nbase = wave * 64;

    const int row = mbase + l15;          // A-fragment row for this lane
    const float dr = dinv[row];
    const float* aggRow  = agg  + (size_t)row * 256;
    const float* selfRow = self + (size_t)row * 256;

    v8f acc[4] = {};

    #pragma unroll 4
    for (int k = 0; k < K_FUSED; k += 32) {
        const int kk = k + half * 16;     // 16-elem run never crosses the 256 split
        const float* ap;
        float scale;
        if (kk < 256) { ap = aggRow + kk;          scale = dr;   }
        else          { ap = selfRow + (kk - 256); scale = 1.0f; }

        v16bf a;
        #pragma unroll
        for (int i = 0; i < 16; ++i) a[i] = (__bf16)(ap[i] * scale);

        #pragma unroll
        for (int t = 0; t < 4; ++t) {
            const int col = nbase + t * 16 + l15;
            const v16bf b = *(const v16bf*)(Bt + (size_t)col * 512 + kk);
            acc[t] = __builtin_amdgcn_wmma_f32_16x16x32_bf16(
                false, a, false, b, (short)0, acc[t], false, false);
        }
    }

    // Epilogue: C/D layout -> VGPR r, lane l: M = r + 8*(l>=16), N = l&15
    #pragma unroll
    for (int t = 0; t < 4; ++t) {
        const int col = nbase + t * 16 + l15;
        const float bv = bias[col];
        #pragma unroll
        for (int r = 0; r < 8; ++r) {
            const int mrow = mbase + r + half * 8;
            out[(size_t)mrow * N + col] = fmaxf(acc[t][r] + bv, 0.0f);
        }
    }
}

// ---------------- classifier + log-softmax (N_CLS = 2) ----------------
__global__ void cls_k(const float* __restrict__ h2, const float* __restrict__ Wc,
                      const float* __restrict__ bc, float* __restrict__ out) {
    int n = blockIdx.x * 256 + threadIdx.x;
    if (n >= N_NODES) return;
    const float* h = h2 + (size_t)n * 128;
    float l0 = bc[0], l1 = bc[1];
    #pragma unroll 8
    for (int k = 0; k < 128; ++k) {
        float v = h[k];
        l0 += v * Wc[k * 2 + 0];
        l1 += v * Wc[k * 2 + 1];
    }
    float m = fmaxf(l0, l1);
    float lse = m + __logf(__expf(l0 - m) + __expf(l1 - m));
    out[(size_t)n * 2 + 0] = l0 - lse;
    out[(size_t)n * 2 + 1] = l1 - lse;
}

// ---------------- launcher ----------------
extern "C" void kernel_launch(void* const* d_in, const int* in_sizes, int n_in,
                              void* d_out, int out_size, void* d_ws, size_t ws_size,
                              hipStream_t stream) {
    const float* x   = (const float*)d_in[0];
    const int*   ei  = (const int*)d_in[1];      // [2][N_EDGES]
    const float* W1l = (const float*)d_in[2];
    const float* W1r = (const float*)d_in[3];
    const float* b1  = (const float*)d_in[4];
    const float* W2l = (const float*)d_in[5];
    const float* W2r = (const float*)d_in[6];
    const float* b2  = (const float*)d_in[7];
    const float* Wc  = (const float*)d_in[8];
    const float* bc  = (const float*)d_in[9];
    float* out = (float*)d_out;

    const int* src = ei;
    const int* dst = ei + N_EDGES;

    float* ws   = (float*)d_ws;
    float* cnt  = ws + O_CNT;
    float* dinv = ws + O_DINV;
    float* agg  = ws + O_AGG;
    float* h1   = ws + O_H1;
    float* h2   = ws + O_H2;
    __bf16* W1t = (__bf16*)(ws + O_W1T);
    __bf16* W2t = (__bf16*)(ws + O_W2T);

    const size_t aggBytes = (size_t)N_NODES * 256 * sizeof(float);

    // degree / mean normalizer
    hipMemsetAsync(cnt, 0, (size_t)N_NODES * sizeof(float), stream);
    deg_count_k<<<(N_EDGES + 255) / 256, 256, 0, stream>>>(dst, cnt);
    deg_inv_k<<<(N_NODES + 255) / 256, 256, 0, stream>>>(cnt, dinv);

    // weights -> bf16, transposed, fused [Wl;Wr]
    prep_w_k<<<(256 * 512 + 128 * 512) / 256, 256, 0, stream>>>(W1l, W1r, W2l, W2r, W1t, W2t);

    // ---- layer 1 ----
    hipMemsetAsync(agg, 0, aggBytes, stream);
    scatter_add_k<<<(N_EDGES * 64) / 256, 256, 0, stream>>>(x, src, dst, agg);
    sage_gemm_k<4><<<N_NODES / 16, 128, 0, stream>>>(agg, x, dinv, W1t, b1, h1, HID);

    // ---- layer 2 ----
    hipMemsetAsync(agg, 0, aggBytes, stream);
    scatter_add_k<<<(N_EDGES * 64) / 256, 256, 0, stream>>>(h1, src, dst, agg);
    sage_gemm_k<2><<<N_NODES / 16, 64, 0, stream>>>(agg, h1, dinv, W2t, b2, h2, HID2);

    // ---- classifier ----
    cls_k<<<(N_NODES + 255) / 256, 256, 0, stream>>>(h2, Wc, bc, out);

    (void)in_sizes; (void)n_in; (void)out_size; (void)ws_size;
}
